// LaterSynthesisBlock_61220463837352
// MI455X (gfx1250) — compile-verified, matrix operand order checked by reading, not run on hardware
//
#include <hip/hip_runtime.h>
#include <cmath>

typedef __attribute__((ext_vector_type(16))) _Float16 v16h;
typedef __attribute__((ext_vector_type(8)))  _Float16 v8h;
typedef __attribute__((ext_vector_type(8)))  float    v8f;

#define LRELU_K 0.2f
#define EPS_IN  1e-5f

__device__ __forceinline__ v16h cat8(v8h lo, v8h hi) {
    return __builtin_shufflevector(lo, hi, 0,1,2,3,4,5,6,7,8,9,10,11,12,13,14,15);
}

__device__ __forceinline__ v8f wmma_f16(v16h a, v16h b, v8f c) {
    return __builtin_amdgcn_wmma_f32_16x16x32_f16(false, a, false, b,
                                                  (short)0, c, false, false);
}

// ---------------------------------------------------------------------------
// Implicit-GEMM 3x3 conv via v_wmma_f32_16x16x32_f16.
//   M (WMMA rows) = output channels (COUT = 256)
//   N (WMMA cols) = pixels (B*64*64 = 65536)
//   K             = ci, staged 64 per barrier, within each of the 9 taps
// Block: 256 thr = 8 waves; block tile = 64 co x 128 px.
// Wave tile: 32 co x 32 px -> 2 A-frags x 2 B-frags -> 8 WMMA per barrier.
// act: f16, UP=true: (B,CIN,32,32) nearest-2x fused; else (B,CIN,64,64).
// wgt: f16, preconverted to (256, 9, CIN) so k is contiguous.
// ---------------------------------------------------------------------------
template<int CIN, bool UP>
__global__ __launch_bounds__(256)
void conv3x3_wmma_kernel(const _Float16* __restrict__ act,
                         const _Float16* __restrict__ wgt,
                         const float* __restrict__ bias,
                         float scale,
                         float* __restrict__ out)   // (B,256,64,64) f32
{
    __shared__ alignas(16) _Float16 Ws[64][72];    // 64 co x 64 k (pad->72)
    __shared__ alignas(16) _Float16 Xs[128][72];   // 128 px x 64 k

    const int tid  = threadIdx.x;
    const int P0   = blockIdx.x * 128;  // pixel tile base
    const int C0   = blockIdx.y * 64;   // cout tile base

    const int wv   = tid >> 5;
    const int lane = tid & 31;
    const int wc   = wv & 1;            // co group of 32
    const int wp   = wv >> 1;           // px group of 32 (0..3)
    const int r    = lane & 15;
    const int hi   = lane >> 4;
    const int kb   = hi << 3;           // per ISA 16-bit A/B lane K-grouping

    // staging maps
    const int wrow = tid >> 2;          // 0..63  (co row)
    const int wkg  = (tid & 3) << 4;    // 0,16,32,48
    const int xrow = tid >> 1;          // 0..127 (px row)
    const int xkg  = (tid & 1) << 5;    // 0,32

    const int px = P0 + xrow;
    const int pb = px >> 12;            // batch
    const int ph = (px >> 6) & 63;
    const int pw = px & 63;

    v8f acc[2][2] = {};

    const int KCH = CIN / 64;           // 64-wide K stages per tap
    for (int t = 0; t < 9 * KCH; ++t) {
        const int tap = t / KCH;
        const int kh  = tap / 3;
        const int kw  = tap - kh * 3;
        const int ci0 = (t - tap * KCH) * 64;

        // ---- stage weights: contiguous f16, 2x b128 per thread ----
        {
            const _Float16* wsrc =
                wgt + ((size_t)(C0 + wrow) * 9 + tap) * CIN + ci0 + wkg;
            *(v8h*)&Ws[wrow][wkg]     = *(const v8h*)(wsrc);
            *(v8h*)&Ws[wrow][wkg + 8] = *(const v8h*)(wsrc + 8);
        }
        // ---- stage activations: 32 f16 per thread, plane-strided gather ----
        {
            const int ih = ph + kh - 1;
            const int iw = pw + kw - 1;
            const bool ok = ((unsigned)ih < 64u) && ((unsigned)iw < 64u);
            const int ihc = ih < 0 ? 0 : (ih > 63 ? 63 : ih);
            const int iwc = iw < 0 ? 0 : (iw > 63 ? 63 : iw);
            size_t base, pstride;
            if (UP) {
                base = (((size_t)pb * CIN + ci0 + xkg) << 10)
                     + ((size_t)(ihc >> 1) << 5) + (iwc >> 1);
                pstride = 1024;
            } else {
                base = (((size_t)pb * CIN + ci0 + xkg) << 12)
                     + ((size_t)ihc << 6) + iwc;
                pstride = 4096;
            }
            #pragma unroll
            for (int g = 0; g < 4; ++g) {
                v8h tmp;
                #pragma unroll
                for (int j = 0; j < 8; ++j) {
                    _Float16 v = act[base + (size_t)(g * 8 + j) * pstride];
                    tmp[j] = ok ? v : (_Float16)0.0f;
                }
                *(v8h*)&Xs[xrow][xkg + g * 8] = tmp;
            }
        }
        __syncthreads();

        // ---- 2 K-steps x (2 A x 2 B) = 8 WMMAs per barrier ----
        #pragma unroll
        for (int ks = 0; ks < 2; ++ks) {
            const int k0 = ks * 32;
            const int ar = wc * 32 + r;
            v16h a0 = cat8(*(const v8h*)&Ws[ar][k0 + kb],
                           *(const v8h*)&Ws[ar][k0 + 16 + kb]);
            v16h a1 = cat8(*(const v8h*)&Ws[ar + 16][k0 + kb],
                           *(const v8h*)&Ws[ar + 16][k0 + 16 + kb]);
            const int br = wp * 32 + r;
            v16h b0 = cat8(*(const v8h*)&Xs[br][k0 + kb],
                           *(const v8h*)&Xs[br][k0 + 16 + kb]);
            v16h b1 = cat8(*(const v8h*)&Xs[br + 16][k0 + kb],
                           *(const v8h*)&Xs[br + 16][k0 + 16 + kb]);
            acc[0][0] = wmma_f16(a0, b0, acc[0][0]);
            acc[0][1] = wmma_f16(a0, b1, acc[0][1]);
            acc[1][0] = wmma_f16(a1, b0, acc[1][0]);
            acc[1][1] = wmma_f16(a1, b1, acc[1][1]);
        }
        __syncthreads();
    }

    // ---- store: D vgpr v, lane<16 -> (M=v,N=lane); lane>=16 -> (M=v+8,N=lane-16)
    #pragma unroll
    for (int ia = 0; ia < 2; ++ia) {
        const int coB = C0 + wc * 32 + ia * 16 + hi * 8;
        #pragma unroll
        for (int ib = 0; ib < 2; ++ib) {
            const int p  = P0 + wp * 32 + ib * 16 + r;
            const int bI = p >> 12;
            const int hw = p & 4095;
            float* o = out + (((size_t)bI * 256) << 12) + hw;
            #pragma unroll
            for (int v = 0; v < 8; ++v) {
                const int co = coB + v;
                o[(size_t)co << 12] = acc[ia][ib][v] * scale + bias[co];
            }
        }
    }
}

// ---------------------------------------------------------------------------
// Weight preconvert: (256, CIN, 3, 3) f32 -> (256, 9, CIN) f16
// ---------------------------------------------------------------------------
__global__ __launch_bounds__(256)
void wcvt_kernel(const float* __restrict__ w, _Float16* __restrict__ wh, int cin)
{
    const int idx = blockIdx.x * 256 + threadIdx.x;   // wh linear index
    const int ci  = idx % cin;
    const int rem = idx / cin;
    const int tap = rem % 9;
    const int co  = rem / 9;
    wh[idx] = (_Float16)w[((size_t)co * cin + ci) * 9 + tap];
}

// ---------------------------------------------------------------------------
// Activation preconvert: f32 -> f16 flat copy
// ---------------------------------------------------------------------------
__global__ __launch_bounds__(256)
void xcvt_kernel(const float* __restrict__ x, _Float16* __restrict__ xh)
{
    const int idx = blockIdx.x * 256 + threadIdx.x;
    xh[idx] = (_Float16)x[idx];
}

// ---------------------------------------------------------------------------
// style[layer][b][j] = (dlat[b, 8+layer, :] . sw[j, :]) * lin + sb[j]
// ---------------------------------------------------------------------------
__global__ __launch_bounds__(256)
void styles_kernel(const float* __restrict__ dlat,
                   const float* __restrict__ sw0, const float* __restrict__ sb0,
                   const float* __restrict__ sw1, const float* __restrict__ sb1,
                   float lin, float* __restrict__ style)
{
    const int idx   = blockIdx.x * 256 + threadIdx.x;  // 0..16383
    const int layer = idx >> 13;
    const int b     = (idx >> 9) & 15;
    const int j     = idx & 511;
    const float* dl = dlat + ((size_t)b * 10 + 8 + layer) * 512;
    const float* sw = (layer ? sw1 : sw0) + (size_t)j * 512;
    const float* sb = layer ? sb1 : sb0;
    float acc = 0.f;
    for (int d = 0; d < 512; ++d) acc += dl[d] * sw[d];
    style[idx] = acc * lin + sb[j];
}

// ---------------------------------------------------------------------------
// blur(1-2-1) + noise + lrelu + instance-norm + style; one block per (b,c).
// Output f16 (feeds conv1 WMMA directly).
// ---------------------------------------------------------------------------
__global__ __launch_bounds__(256)
void blur_epilogue0_kernel(const float* __restrict__ y0,
                           const float* __restrict__ noise,
                           const float* __restrict__ nw,
                           const float* __restrict__ style,   // (B,512) layer0
                           _Float16* __restrict__ y1)
{
    __shared__ float tile[66][66];
    __shared__ float reds[256];
    __shared__ float redq[256];

    const int bc  = blockIdx.x;
    const int c   = bc & 255;
    const int b   = bc >> 8;
    const int tid = threadIdx.x;
    const float* src = y0 + ((size_t)bc << 12);

    for (int i = tid; i < 66 * 66; i += 256) {
        const int rr = i / 66;
        const int cc = i - rr * 66;
        float v = 0.0f;
        if (rr >= 1 && rr <= 64 && cc >= 1 && cc <= 64)
            v = src[((rr - 1) << 6) + (cc - 1)];
        tile[rr][cc] = v;
    }
    __syncthreads();

    const float nwc = nw[c];
    const float* nz = noise + ((size_t)b << 12);
    float lv[16];
    float s = 0.f, sq = 0.f;
    #pragma unroll
    for (int i = 0; i < 16; ++i) {
        const int p = tid + i * 256;
        const int h = p >> 6;
        const int w = p & 63;
        float bsum =      tile[h  ][w] + 2.f*tile[h  ][w+1] +     tile[h  ][w+2]
                   + 2.f*tile[h+1][w] + 4.f*tile[h+1][w+1] + 2.f*tile[h+1][w+2]
                   +      tile[h+2][w] + 2.f*tile[h+2][w+1] +     tile[h+2][w+2];
        float v = bsum * (1.f / 16.f) + nwc * nz[p];
        v = v >= 0.f ? v : LRELU_K * v;
        lv[i] = v;
        s += v; sq += v * v;
    }
    reds[tid] = s; redq[tid] = sq;
    __syncthreads();
    for (int st = 128; st > 0; st >>= 1) {
        if (tid < st) { reds[tid] += reds[tid + st]; redq[tid] += redq[tid + st]; }
        __syncthreads();
    }
    const float mean = reds[0] * (1.f / 4096.f);
    const float var  = redq[0] * (1.f / 4096.f) - mean * mean;
    const float inv  = rsqrtf(var + EPS_IN);
    const float ys   = style[(size_t)b * 512 + c] + 1.0f;
    const float yb   = style[(size_t)b * 512 + 256 + c];
    _Float16* dst = y1 + ((size_t)bc << 12);
    #pragma unroll
    for (int i = 0; i < 16; ++i) {
        const int p = tid + i * 256;
        dst[p] = (_Float16)(((lv[i] - mean) * inv) * ys + yb);
    }
}

// ---------------------------------------------------------------------------
// noise + lrelu + instance-norm + style, in place on d_out (f32).
// ---------------------------------------------------------------------------
__global__ __launch_bounds__(256)
void epilogue1_kernel(float* __restrict__ y,
                      const float* __restrict__ noise,
                      const float* __restrict__ nw,
                      const float* __restrict__ style)   // (B,512) layer1
{
    __shared__ float reds[256];
    __shared__ float redq[256];

    const int bc  = blockIdx.x;
    const int c   = bc & 255;
    const int b   = bc >> 8;
    const int tid = threadIdx.x;
    float* p = y + ((size_t)bc << 12);
    const float* nz = noise + ((size_t)b << 12);
    const float nwc = nw[c];

    float lv[16];
    float s = 0.f, sq = 0.f;
    #pragma unroll
    for (int i = 0; i < 16; ++i) {
        const int px = tid + i * 256;
        float v = p[px] + nwc * nz[px];
        v = v >= 0.f ? v : LRELU_K * v;
        lv[i] = v;
        s += v; sq += v * v;
    }
    reds[tid] = s; redq[tid] = sq;
    __syncthreads();
    for (int st = 128; st > 0; st >>= 1) {
        if (tid < st) { reds[tid] += reds[tid + st]; redq[tid] += redq[tid + st]; }
        __syncthreads();
    }
    const float mean = reds[0] * (1.f / 4096.f);
    const float var  = redq[0] * (1.f / 4096.f) - mean * mean;
    const float inv  = rsqrtf(var + EPS_IN);
    const float ys   = style[(size_t)b * 512 + c] + 1.0f;
    const float yb   = style[(size_t)b * 512 + 256 + c];
    #pragma unroll
    for (int i = 0; i < 16; ++i) {
        const int px = tid + i * 256;
        p[px] = ((lv[i] - mean) * inv) * ys + yb;
    }
}

// ---------------------------------------------------------------------------
extern "C" void kernel_launch(void* const* d_in, const int* in_sizes, int n_in,
                              void* d_out, int out_size, void* d_ws, size_t ws_size,
                              hipStream_t stream) {
    const float* x    = (const float*)d_in[0];   // (16,512,32,32)
    const float* dlat = (const float*)d_in[1];   // (16,10,512)
    const float* n0   = (const float*)d_in[2];   // (16,1,64,64)
    const float* n1   = (const float*)d_in[3];
    const float* w0   = (const float*)d_in[4];   // (256,512,3,3)
    const float* b0   = (const float*)d_in[5];
    const float* nw0  = (const float*)d_in[6];
    const float* sw0  = (const float*)d_in[7];   // (512,512)
    const float* sb0  = (const float*)d_in[8];
    const float* w1   = (const float*)d_in[9];   // (256,256,3,3)
    const float* b1   = (const float*)d_in[10];
    const float* nw1  = (const float*)d_in[11];
    const float* sw1  = (const float*)d_in[12];
    const float* sb1  = (const float*)d_in[13];
    float* out = (float*)d_out;                  // (16,256,64,64)

    // workspace layout
    char* ws = (char*)d_ws;
    float*    style = (float*)ws;                          // 64 KB
    _Float16* y1    = (_Float16*)(ws + (2*16*512*4));      // 33.55 MB
    _Float16* xh    = (_Float16*)(ws + 33619968);          // 16.78 MB
    _Float16* w0h   = (_Float16*)(ws + 50397184);          // 2.36 MB
    _Float16* w1h   = (_Float16*)(ws + 52756480);          // 1.18 MB

    const float lin = 1.0f / sqrtf(512.0f);
    const float s0  = sqrtf(2.0f) / sqrtf(512.0f * 9.0f);
    const float s1  = sqrtf(2.0f) / sqrtf(256.0f * 9.0f);

    // 0) precision/layout preconversion (one pass each, trivial bandwidth)
    xcvt_kernel<<<32768, 256, 0, stream>>>(x, xh);                 // 16*512*1024
    wcvt_kernel<<<4608, 256, 0, stream>>>(w0, w0h, 512);           // 256*9*512
    wcvt_kernel<<<2304, 256, 0, stream>>>(w1, w1h, 256);           // 256*9*256
    // 1) style linears for both layers
    styles_kernel<<<64, 256, 0, stream>>>(dlat, sw0, sb0, sw1, sb1, lin, style);
    // 2) conv0 (fused 2x nearest upsample) -> d_out (f32 staging)
    conv3x3_wmma_kernel<512, true><<<dim3(512, 4), 256, 0, stream>>>(
        xh, w0h, b0, s0, out);
    // 3) blur + epilogue0 -> y1 (f16)
    blur_epilogue0_kernel<<<4096, 256, 0, stream>>>(out, n0, nw0, style, y1);
    // 4) conv1 -> d_out (f32)
    conv3x3_wmma_kernel<256, false><<<dim3(512, 4), 256, 0, stream>>>(
        y1, w1h, b1, s1, out);
    // 5) epilogue1 in place on d_out
    epilogue1_kernel<<<4096, 256, 0, stream>>>(out, n1, nw1, style + 16 * 512);
}
